// PowerFlowEquation_16355235463758
// MI455X (gfx1250) — matrix operand order repaired and checked by reading, not compiled, not run on hardware
//
#include <hip/hip_runtime.h>

#define NN 4096
#define BB 64
#define TT (BB * NN)          // 262144 elements per [B,N] plane
#define KSPLITS 8
#define KCH (NN / KSPLITS)    // 512 k per split

typedef __attribute__((ext_vector_type(16))) __bf16 v16bf;
typedef __attribute__((ext_vector_type(8)))  __bf16 v8bf;
typedef __attribute__((ext_vector_type(8)))  float  v8f;
typedef __attribute__((ext_vector_type(16))) unsigned short v16us;

// ---- WMMA wrapper: D = A(16x32 bf16) * B(32x16 bf16) + C(16x16 f32) ----
static __device__ __forceinline__ v8f wmma_bf16(v16bf a, v16bf b, v8f c) {
    return __builtin_amdgcn_wmma_f32_16x16x32_bf16(
        /*neg_a=*/false, a, /*neg_b=*/false, b,
        /*c_mod=*/(short)0, c, /*reuse_a=*/false, /*reuse_b=*/false);
}

// Exact bf16 negation (flip sign bit of all 16 packed elements)
static __device__ __forceinline__ v16bf negfrag(v16bf x) {
    v16us u = __builtin_bit_cast(v16us, x);
    u = u ^ (unsigned short)0x8000;
    return __builtin_bit_cast(v16bf, u);
}

// B-operand fragment (K x N = 32 x 16 bf16, wave32):
//   lanes 0-15 : N = lane,    K = k0..k0+15   (VGPR v holds K=2v,2v+1)
//   lanes16-31 : N = lane-16, K = k0+16..k0+31
// Per lane: 16 consecutive f32 from one G row -> split into hi/lo bf16.
static __device__ __forceinline__ void load_mat_frag(const float* __restrict__ p,
                                                     v16bf& hi, v16bf& lo) {
    const float4* p4 = (const float4*)p;
    float4 a = p4[0], b = p4[1], c = p4[2], d = p4[3];
    float x[16];
    x[0]=a.x;  x[1]=a.y;  x[2]=a.z;  x[3]=a.w;
    x[4]=b.x;  x[5]=b.y;  x[6]=b.z;  x[7]=b.w;
    x[8]=c.x;  x[9]=c.y;  x[10]=c.z; x[11]=c.w;
    x[12]=d.x; x[13]=d.y; x[14]=d.z; x[15]=d.w;
#pragma unroll
    for (int e = 0; e < 16; ++e) {
        __bf16 h = (__bf16)x[e];
        hi[e] = h;
        lo[e] = (__bf16)(x[e] - (float)h);
    }
}

// A-operand fragment (M x K = 16 x 32 bf16, wave32):
//   lanes 0-15 : M = lane,    e0..e15 = K 0..15
//   lanes16-31 : M = lane-16, e0..e15 = K 16..31
// p points at chunk0 (8 bf16); chunk1 is at p+8.
static __device__ __forceinline__ v16bf load_vec_frag(const __bf16* __restrict__ p) {
    v8bf a = *(const v8bf*)p;
    v8bf b = *(const v8bf*)(p + 8);
    v16bf r;
#pragma unroll
    for (int i = 0; i < 8; ++i) { r[i] = a[i]; r[i + 8] = b[i]; }
    return r;
}

// ---- Kernel 1: trig prep -> c,s (f32) and hi/lo bf16 splits ----
__global__ void pf_prep(const float* __restrict__ Vm, const float* __restrict__ Va,
                        float* __restrict__ cF, float* __restrict__ sF,
                        __bf16* __restrict__ chi, __bf16* __restrict__ clo,
                        __bf16* __restrict__ shi, __bf16* __restrict__ slo) {
    int idx = blockIdx.x * blockDim.x + threadIdx.x;
    if (idx >= TT) return;
    float vm = Vm[idx], a = Va[idx];
    float co = cosf(a), si = sinf(a);
    float c = vm * co, s = vm * si;
    cF[idx] = c; sF[idx] = s;
    __bf16 ch = (__bf16)c;
    chi[idx] = ch; clo[idx] = (__bf16)(c - (float)ch);
    __bf16 sh = (__bf16)s;
    shi[idx] = sh; slo[idx] = (__bf16)(s - (float)sh);
}

// ---- Kernel 2: WMMA GEMM.  X = G*c + B*s,  Y = G*s + (-B)*c  (per k-split partials) ----
// block = 128 threads (4 waves); wave w owns i-tile (blockIdx.x*4 + w); blockIdx.y = k-split.
__global__ void __launch_bounds__(128)
pf_wmma_gemm(const float* __restrict__ G, const float* __restrict__ Bm,
             const __bf16* __restrict__ chi, const __bf16* __restrict__ clo,
             const __bf16* __restrict__ shi, const __bf16* __restrict__ slo,
             float* __restrict__ Xp, float* __restrict__ Yp) {
    const int lane = threadIdx.x & 31;
    const int wave = threadIdx.x >> 5;
    const int half = lane >> 4;
    const int lrow = lane & 15;
    const int ibase = (blockIdx.x * 4 + wave) * 16;   // bus-index tile base
    const int split = blockIdx.y;
    const int kbeg  = split * KCH;

    v8f accX[4], accY[4];
#pragma unroll
    for (int t = 0; t < 4; ++t)
#pragma unroll
        for (int v = 0; v < 8; ++v) { accX[t][v] = 0.0f; accY[t][v] = 0.0f; }

    const size_t mrow = (size_t)(ibase + lrow) * NN + (size_t)(half * 16);
    const float* gp = G  + mrow;
    const float* bp = Bm + mrow;

    for (int k0 = kbeg; k0 < kbeg + KCH; k0 += 32) {
        v16bf ghi, glo, bhi, blo;
        load_mat_frag(gp + k0, ghi, glo);
        load_mat_frag(bp + k0, bhi, blo);
        // Negate B-matrix once per k-chunk (16 XORs) instead of the c-vector
        // fragments per batch tile (64 XORs):  Y += G*s + (-B)*c
        const v16bf nbhi = negfrag(bhi);
        const v16bf nblo = negfrag(blo);

#pragma unroll
        for (int t = 0; t < 4; ++t) {
            const int off = (t * 16 + lrow) * NN + k0 + half * 16;
            v16bf vchi = load_vec_frag(chi + off);
            v16bf vclo = load_vec_frag(clo + off);
            v16bf vshi = load_vec_frag(shi + off);
            v16bf vslo = load_vec_frag(slo + off);

            // X += G*c (bf16x3) + B*s (bf16x3)
            accX[t] = wmma_bf16(vchi, ghi,  accX[t]);
            accX[t] = wmma_bf16(vclo, ghi,  accX[t]);
            accX[t] = wmma_bf16(vchi, glo,  accX[t]);
            accX[t] = wmma_bf16(vshi, bhi,  accX[t]);
            accX[t] = wmma_bf16(vslo, bhi,  accX[t]);
            accX[t] = wmma_bf16(vshi, blo,  accX[t]);
            // Y += G*s (bf16x3) + (-B)*c (bf16x3)
            accY[t] = wmma_bf16(vshi, ghi,  accY[t]);
            accY[t] = wmma_bf16(vslo, ghi,  accY[t]);
            accY[t] = wmma_bf16(vshi, glo,  accY[t]);
            accY[t] = wmma_bf16(vchi, nbhi, accY[t]);
            accY[t] = wmma_bf16(vclo, nbhi, accY[t]);
            accY[t] = wmma_bf16(vchi, nblo, accY[t]);
        }
    }

    // C/D layout: lane l -> N = l%16; VGPR v -> M = v + 8*(l/16)
    float* xo = Xp + (size_t)split * TT;
    float* yo = Yp + (size_t)split * TT;
    const int icol = ibase + lrow;
#pragma unroll
    for (int t = 0; t < 4; ++t)
#pragma unroll
        for (int v = 0; v < 8; ++v) {
            const int brow = t * 16 + v + 8 * half;
            xo[(size_t)brow * NN + icol] = accX[t][v];
            yo[(size_t)brow * NN + icol] = accY[t][v];
        }
}

// ---- Kernel 3: reduce k-split partials + residuals ----
__global__ void pf_final(const float* __restrict__ cF, const float* __restrict__ sF,
                         const float* __restrict__ Xp, const float* __restrict__ Yp,
                         const float* __restrict__ Pin, const float* __restrict__ Qin,
                         float* __restrict__ out) {
    int idx = blockIdx.x * blockDim.x + threadIdx.x;
    if (idx >= TT) return;
    float X = 0.0f, Y = 0.0f;
#pragma unroll
    for (int sp = 0; sp < KSPLITS; ++sp) {
        X += Xp[(size_t)sp * TT + idx];
        Y += Yp[(size_t)sp * TT + idx];
    }
    float c = cF[idx], s = sF[idx];
    out[idx]      = c * X + s * Y - Pin[idx];   // res_P
    out[TT + idx] = s * X - c * Y - Qin[idx];   // res_Q
}

extern "C" void kernel_launch(void* const* d_in, const int* in_sizes, int n_in,
                              void* d_out, int out_size, void* d_ws, size_t ws_size,
                              hipStream_t stream) {
    const float* Vm  = (const float*)d_in[0];
    const float* Va  = (const float*)d_in[1];
    const float* Pin = (const float*)d_in[2];
    const float* Qin = (const float*)d_in[3];
    const float* G   = (const float*)d_in[4];
    const float* Bm  = (const float*)d_in[5];

    char* ws = (char*)d_ws;
    float*  cF  = (float*)ws;  ws += (size_t)TT * sizeof(float);
    float*  sF  = (float*)ws;  ws += (size_t)TT * sizeof(float);
    __bf16* chi = (__bf16*)ws; ws += (size_t)TT * sizeof(__bf16);
    __bf16* clo = (__bf16*)ws; ws += (size_t)TT * sizeof(__bf16);
    __bf16* shi = (__bf16*)ws; ws += (size_t)TT * sizeof(__bf16);
    __bf16* slo = (__bf16*)ws; ws += (size_t)TT * sizeof(__bf16);
    float*  Xp  = (float*)ws;  ws += (size_t)KSPLITS * TT * sizeof(float);
    float*  Yp  = (float*)ws;

    pf_prep<<<TT / 256, 256, 0, stream>>>(Vm, Va, cF, sF, chi, clo, shi, slo);

    // N/16 = 256 i-tiles; 4 waves per block -> 64 blocks; 8 k-splits -> 2048 waves
    pf_wmma_gemm<<<dim3(NN / 64, KSPLITS), 128, 0, stream>>>(
        G, Bm, chi, clo, shi, slo, Xp, Yp);

    pf_final<<<TT / 256, 256, 0, stream>>>(cF, sF, Xp, Yp, Pin, Qin, (float*)d_out);
}
